// DistanceTransform_84241488544210
// MI455X (gfx1250) — matrix operand order repaired and checked by reading, not compiled, not run on hardware
//
#include <hip/hip_runtime.h>

typedef __attribute__((ext_vector_type(16))) _Float16 v16h;
typedef __attribute__((ext_vector_type(8)))  _Float16 v8h;
typedef __attribute__((ext_vector_type(8)))  float    v8f;

#define IMG_H   384
#define IMG_W   384
#define NPLANE  6            // B*C = 2*3
#define HPAR    0.35f
#define HLN2    0.24260226f  // HPAR * ln(2): -h*ln(x) = -HLN2*log2(x)
#define NITER   128          // ceil(384 / 3)
#define LDS_W   64           // LDS row stride in halves (128B): shift addressing
#define LDS_H   40           // 3 + 32 + 3 halo rows, padded
#define BFRAG_N (7 * 32 * 16)  // 7 dy-fragments, 32 lanes x 16 halves each

// ---------------------------------------------------------------------------
// init: boundary(f16) <- image(f32), out <- 0  (harness poisons d_out/d_ws)
// ---------------------------------------------------------------------------
__global__ void cdt_init(const float* __restrict__ img,
                         _Float16* __restrict__ bnd,
                         float* __restrict__ out, int n) {
  int i = blockIdx.x * blockDim.x + threadIdx.x;
  if (i < n) {
    bnd[i] = (_Float16)img[i];
    out[i] = 0.0f;
  }
}

// ---------------------------------------------------------------------------
// Precompute the 7 Toeplitz B fragments in WMMA per-lane register layout:
// bfrag[((dy*32 + lane)*16) + j] = B_dy[k, n],  k = (j>=8?16:0)+(lane>=16?8:0)+(j&7),
// n = lane&15, B_dy[k,n] = exp(-hypot(dy-3, (k-n)-3)/h) for k-n in [0,6], else 0.
// Each lane then loads its v16h B with two aligned ds_load_b128 — no branches.
// ---------------------------------------------------------------------------
__global__ void build_bfrag(_Float16* __restrict__ bfrag) {
  int idx = blockIdx.x * blockDim.x + threadIdx.x;
  if (idx >= BFRAG_N) return;
  int dy   = idx >> 9;            // /512
  int lane = (idx >> 4) & 31;
  int j    = idx & 15;
  int n    = lane & 15;
  int k    = ((j >> 3) << 4) + ((lane >> 4) << 3) + (j & 7);
  int diff = k - n;               // dx index into kernel row
  float dyy = (float)(dy - 3);
  float dxx = (float)(diff - 3);
  float w = __expf(-sqrtf(dyy * dyy + dxx * dxx) * (1.0f / HPAR));
  bfrag[idx] = (diff >= 0 && diff < 7) ? (_Float16)w : (_Float16)0.0f;
}

// ---------------------------------------------------------------------------
// One CDT iteration. Block = 128 threads (4 wave32), covers a 32x32 output
// region; each wave computes one 16x16 tile via 7 accumulating WMMAs
// (one per kernel row dy), conv lowered to Toeplitz-matrix GEMMs.
// ---------------------------------------------------------------------------
__global__ __launch_bounds__(128) void cdt_iter(const _Float16* __restrict__ bin,
                                                const _Float16* __restrict__ bfrag,
                                                _Float16* __restrict__ bout,
                                                float* __restrict__ out,
                                                float offs) {
  __shared__ __align__(16) _Float16 tile[LDS_H * LDS_W];
  __shared__ __align__(16) _Float16 bfr[BFRAG_N];

  const int tid   = threadIdx.x;
  const int plane = blockIdx.z;
  const int by    = blockIdx.y * 32;
  const int bx    = blockIdx.x * 32;
  const _Float16* src = bin + (size_t)plane * (IMG_H * IMG_W);

  // Copy precomputed B fragments into LDS (vectorized, 448 x v8h chunks).
  {
    const v8h* gsrc = (const v8h*)bfrag;
    v8h* ldst = (v8h*)bfr;
    for (int i = tid; i < BFRAG_N / 8; i += 128) ldst[i] = gsrc[i];
  }

  // Stage input tile with replicate (edge) padding. LDS col 0 == global x-3.
  for (int idx = tid; idx < LDS_H * LDS_W; idx += 128) {
    int r = idx >> 6, c = idx & 63;
    int gy = min(max(by - 3 + r, 0), IMG_H - 1);
    int gx = min(max(bx - 3 + c, 0), IMG_W - 1);
    tile[idx] = src[gy * IMG_W + gx];
  }
  __syncthreads();

  const int wave = tid >> 5;
  const int lane = tid & 31;
  const int g    = lane >> 4;          // lane group (K-half select)
  const int nn   = lane & 15;          // A row / B-C-D column
  const int ty   = (wave >> 1) * 16;   // wave tile origin in block region
  const int tx   = (wave & 1) * 16;

  v8f c = {};                          // f32 accumulator, 16x16 tile
#pragma unroll
  for (int dy = 0; dy < 7; ++dy) {
    // ---- A fragment: 16x32 f16. Lane layout (ISA 7.12.2): row M = lane%16;
    // lanes<16 hold K{0..7,16..23}, lanes>=16 hold K{8..15,24..31}.
    // With LDS col 0 at x-3, those are two 16B-aligned 8-half runs.
    int arow = ty + nn + dy;                  // input row index in LDS
    int acol = tx + g * 8;                    // start of first K run
    const v8h* ap = (const v8h*)&tile[(arow << 6) + acol];
    v8h a1 = ap[0];                           // K = kbaseA .. +7
    v8h a2 = ap[2];                           // K = kbaseA+16 .. +23
    v16h a;
#pragma unroll
    for (int j = 0; j < 8; ++j) { a[j] = a1[j]; a[j + 8] = a2[j]; }

    // ---- B fragment: precomputed per-lane layout, two aligned b128 loads.
    const v8h* bp = (const v8h*)&bfr[((dy * 32 + lane) << 4)];
    v8h b1 = bp[0];
    v8h b2 = bp[1];
    v16h b;
#pragma unroll
    for (int j = 0; j < 8; ++j) { b[j] = b1[j]; b[j + 8] = b2[j]; }

    // D = A*B + C  (accumulate the 7 kernel-row contributions)
    c = __builtin_amdgcn_wmma_f32_16x16x32_f16(false, a, false, b,
                                               (short)0, c, false, false);
  }

  // ---- Epilogue (branchless): cdt = -h*ln(conv) = -(h*ln2)*log2(conv).
  // Select the log operand, not the code path: log2(1)=0 for masked-off lanes,
  // so v_log_f32 runs unconditionally and only v_cndmask selects results.
  const size_t pbase = (size_t)plane * (IMG_H * IMG_W);
#pragma unroll
  for (int r = 0; r < 8; ++r) {
    int my = r + g * 8;                       // C layout: lanes>=16 -> M+8
    int py = by + ty + my;
    int px = bx + tx + nn;
    float conv = c[r];
    _Float16 old = tile[((ty + my + 3) << 6) + (tx + nn + 3)];
    bool mask = (conv > 0.0f) && (conv < 1.0f);     // cdt > 0
    float lg  = __log2f(mask ? conv : 1.0f);        // unconditional TRANS op
    float add = mask ? (offs - HLN2 * lg) : 0.0f;
    _Float16 nb = mask ? (_Float16)1.0f : old;
    size_t idx = pbase + (size_t)py * IMG_W + px;
    out[idx] += add;                          // pixel owned by exactly one lane
    bout[idx] = nb;
  }
}

// ---------------------------------------------------------------------------
extern "C" void kernel_launch(void* const* d_in, const int* in_sizes, int n_in,
                              void* d_out, int out_size, void* d_ws, size_t ws_size,
                              hipStream_t stream) {
  const float* img = (const float*)d_in[0];
  float* out = (float*)d_out;

  const int n = NPLANE * IMG_H * IMG_W;       // 884736
  _Float16* bnd0  = (_Float16*)d_ws;
  _Float16* bnd1  = bnd0 + n;
  _Float16* bfrag = bnd1 + n;                 // + 7 KB of B fragments

  cdt_init<<<(n + 255) / 256, 256, 0, stream>>>(img, bnd0, out, n);
  build_bfrag<<<(BFRAG_N + 255) / 256, 256, 0, stream>>>(bfrag);

  dim3 grid(IMG_W / 32, IMG_H / 32, NPLANE);  // 12 x 12 x 6
  for (int i = 0; i < NITER; ++i) {
    _Float16* bin  = (i & 1) ? bnd1 : bnd0;
    _Float16* bout = (i & 1) ? bnd0 : bnd1;
    cdt_iter<<<grid, dim3(128), 0, stream>>>(bin, bfrag, bout, out, 3.0f * (float)i);
  }
}
